// BinaryDecorator_25426206392945
// MI455X (gfx1250) — 8-bit matrix operand order repaired and checked by reading, compiled, not run on hardware
//
#include <hip/hip_runtime.h>
#include <hip/hip_bf16.h>

typedef int  v8i  __attribute__((ext_vector_type(8)));

static constexpr size_t XB_OFF   = 0;
static constexpr size_t WB_OFF   = (size_t)65536 * 512;
static constexpr size_t PART_OFF = WB_OFF + (size_t)512 * 512;
static constexpr size_t MEAN_OFF = PART_OFF + 2048 * sizeof(float);

static constexpr int Bsz = 65536;
static constexpr int IN  = 512;
static constexpr int OUT = 512;

__global__ void quantize_x_abs_sum(const float* __restrict__ x,
                                   char* __restrict__ xb,
                                   float* __restrict__ partials) {
  const int nvec = (Bsz * IN) / 4;
  float s = 0.f;
  int stride = gridDim.x * blockDim.x;
  int*  xb32 = (int*)xb;
  const float4* x4 = (const float4*)x;
  for (int i = blockIdx.x * blockDim.x + threadIdx.x; i < nvec; i += stride) {
    float4 v = x4[i];
    s += fabsf(v.x) + fabsf(v.y) + fabsf(v.z) + fabsf(v.w);
    int c0 = (v.x > 0.f) - (v.x < 0.f);
    int c1 = (v.y > 0.f) - (v.y < 0.f);
    int c2 = (v.z > 0.f) - (v.z < 0.f);
    int c3 = (v.w > 0.f) - (v.w < 0.f);
    xb32[i] = (c0 & 0xFF) | ((c1 & 0xFF) << 8) | ((c2 & 0xFF) << 16) | ((c3 & 0xFF) << 24);
  }
  __shared__ float red[256];
  red[threadIdx.x] = s;
  __syncthreads();
  for (int off = 128; off > 0; off >>= 1) {
    if (threadIdx.x < off) red[threadIdx.x] += red[threadIdx.x + off];
    __syncthreads();
  }
  if (threadIdx.x == 0) partials[blockIdx.x] = red[0];
}

__global__ void finalize_mean(const float* __restrict__ partials,
                              float* __restrict__ mean_out) {
  float s = 0.f;
  for (int i = threadIdx.x; i < 2048; i += 256) s += partials[i];
  __shared__ float red[256];
  red[threadIdx.x] = s;
  __syncthreads();
  for (int off = 128; off > 0; off >>= 1) {
    if (threadIdx.x < off) red[threadIdx.x] += red[threadIdx.x + off];
    __syncthreads();
  }
  if (threadIdx.x == 0) mean_out[0] = red[0] / (float)((size_t)Bsz * IN);
}

__global__ void quantize_w(const float* __restrict__ W, char* __restrict__ wb) {
  const int nvec = (OUT * IN) / 4;
  int i = blockIdx.x * blockDim.x + threadIdx.x;
  if (i >= nvec) return;
  float4 v = ((const float4*)W)[i];
  int c0 = (v.x > 0.f) - (v.x < 0.f);
  int c1 = (v.y > 0.f) - (v.y < 0.f);
  int c2 = (v.z > 0.f) - (v.z < 0.f);
  int c3 = (v.w > 0.f) - (v.w < 0.f);
  ((int*)wb)[i] = (c0 & 0xFF) | ((c1 & 0xFF) << 8) | ((c2 & 0xFF) << 16) | ((c3 & 0xFF) << 24);
}

__global__ void __launch_bounds__(256)
binary_gemm_wmma(const char* __restrict__ xb,
                 const char* __restrict__ wb,
                 const float* __restrict__ bias,
                 const float* __restrict__ mean_p,
                 float* __restrict__ out) {
  constexpr int LROW = 528;
  __shared__ int4 ldsW4[64 * (LROW / 16)];
  char* lds = (char*)ldsW4;

  const int tid  = threadIdx.x;
  const int wave = tid >> 5;
  const int lane = tid & 31;
  const int lo   = lane & 15;
  const int hi   = lane >> 4;

  const int mBase = blockIdx.x * 256 + wave * 32;
  const int nBase = blockIdx.y * 64;

  {
    const int row = tid >> 2;
    const int grp = tid & 3;
    const int4* src = (const int4*)(wb + (size_t)(nBase + row) * IN + grp * 128);
    int4*       dst = (int4*)(lds + row * LROW + grp * 128);
#pragma unroll
    for (int i = 0; i < 8; ++i) dst[i] = src[i];
  }
  __syncthreads();

  v8i acc0[4], acc1[4];
#pragma unroll
  for (int nt = 0; nt < 4; ++nt) {
    acc0[nt] = (v8i){0, 0, 0, 0, 0, 0, 0, 0};
    acc1[nt] = (v8i){0, 0, 0, 0, 0, 0, 0, 0};
  }

  const char* arow0 = xb + (size_t)(mBase + lo) * IN + (hi << 3);
  const char* arow1 = arow0 + (size_t)16 * IN;

#pragma unroll
  for (int k0 = 0; k0 < IN; k0 += 64) {
    v8i a0, a1;
    {
      const char* p = arow0 + k0;
      int2 p0 = *(const int2*)(p + 0);
      int2 p1 = *(const int2*)(p + 16);
      int2 p2 = *(const int2*)(p + 32);
      int2 p3 = *(const int2*)(p + 48);
      a0[0] = p0.x; a0[1] = p0.y; a0[2] = p1.x; a0[3] = p1.y;
      a0[4] = p2.x; a0[5] = p2.y; a0[6] = p3.x; a0[7] = p3.y;
    }
    {
      const char* p = arow1 + k0;
      int2 p0 = *(const int2*)(p + 0);
      int2 p1 = *(const int2*)(p + 16);
      int2 p2 = *(const int2*)(p + 32);
      int2 p3 = *(const int2*)(p + 48);
      a1[0] = p0.x; a1[1] = p0.y; a1[2] = p1.x; a1[3] = p1.y;
      a1[4] = p2.x; a1[5] = p2.y; a1[6] = p3.x; a1[7] = p3.y;
    }

    v8i bf[4];
#pragma unroll
    for (int nt = 0; nt < 4; ++nt) {
      const char* q = lds + (nt * 16 + lo) * LROW + k0 + (hi << 3);
      int2 q0 = *(const int2*)(q);
      int2 q1 = *(const int2*)(q + 16);
      int2 q2 = *(const int2*)(q + 32);
      int2 q3 = *(const int2*)(q + 48);
      bf[nt][0] = q0.x; bf[nt][1] = q0.y; bf[nt][2] = q1.x; bf[nt][3] = q1.y;
      bf[nt][4] = q2.x; bf[nt][5] = q2.y; bf[nt][6] = q3.x; bf[nt][7] = q3.y;
    }

#pragma unroll
    for (int nt = 0; nt < 4; ++nt) {
      acc0[nt] = __builtin_amdgcn_wmma_i32_16x16x64_iu8(
          true, a0, true, bf[nt], acc0[nt], false, false);
      acc1[nt] = __builtin_amdgcn_wmma_i32_16x16x64_iu8(
          true, a1, true, bf[nt], acc1[nt], false, false);
    }
  }

  const float mean = mean_p[0];
#pragma unroll
  for (int nt = 0; nt < 4; ++nt) {
    const int col = nBase + nt * 16 + lo;
    const float bv = bias[col];
#pragma unroll
    for (int r = 0; r < 8; ++r) {
      const int row0 = mBase + r + hi * 8;
      const int row1 = row0 + 16;
      __builtin_nontemporal_store(((float)acc0[nt][r] + bv) * mean,
                                  &out[(size_t)row0 * OUT + col]);
      __builtin_nontemporal_store(((float)acc1[nt][r] + bv) * mean,
                                  &out[(size_t)row1 * OUT + col]);
    }
  }
}

extern "C" void kernel_launch(void* const* d_in, const int* in_sizes, int n_in,
                              void* d_out, int out_size, void* d_ws, size_t ws_size,
                              hipStream_t stream) {
  const float* x    = (const float*)d_in[0];
  const float* W    = (const float*)d_in[1];
  const float* bias = (const float*)d_in[2];
  float*       out  = (float*)d_out;

  char*  xb       = (char*)d_ws + XB_OFF;
  char*  wb       = (char*)d_ws + WB_OFF;
  float* partials = (float*)((char*)d_ws + PART_OFF);
  float* mean_p   = (float*)((char*)d_ws + MEAN_OFF);

  quantize_x_abs_sum<<<2048, 256, 0, stream>>>(x, xb, partials);
  finalize_mean<<<1, 256, 0, stream>>>(partials, mean_p);
  quantize_w<<<(OUT * IN / 4 + 255) / 256, 256, 0, stream>>>(W, wb);

  dim3 grid(Bsz / 256, OUT / 64);
  binary_gemm_wmma<<<grid, 256, 0, stream>>>(xb, wb, bias, mean_p, out);
}
